// MixtureOfSoftmaxes_62586263437938
// MI455X (gfx1250) — matrix operand order repaired and checked by reading, not compiled, not run on hardware
//
#include <hip/hip_runtime.h>
#include <stdint.h>

// Problem constants (match reference)
#define B_    1024
#define H_    4
#define D_    256
#define V_    100000
#define ROWS_ (B_ * H_)                       // 4096 GEMM rows (b*4 + h)
#define NCHUNK_ 16                            // V-chunks for pass-1 parallelism
#define NITER_  ((V_ + 63) / 64)              // 1563 64-column tiles
#define IPC_    ((NITER_ + NCHUNK_ - 1) / NCHUNK_)  // 98 iters per chunk
#define TPITCH_ 264                           // padded LDS tile row pitch (bf16)
#define TILE_ELEMS_ (64 * TPITCH_)

// CDNA5 WMMA operand types (wave32): 16x16x32 bf16 -> f32
typedef __attribute__((ext_vector_type(16))) __bf16 bf16x16;
typedef __attribute__((ext_vector_type(8)))  float  f32x8;

union BfTile { bf16x16 v; uint4 q[2]; };

// ---------- helpers ----------
__device__ __forceinline__ unsigned short f2bf(float a) {
  unsigned int u = __float_as_uint(a);
  return (unsigned short)((u + 0x7FFFu + ((u >> 16) & 1u)) >> 16);  // RNE
}
__device__ __forceinline__ unsigned int packbf2(float a, float b) {
  return (unsigned int)f2bf(a) | ((unsigned int)f2bf(b) << 16);
}
__device__ __forceinline__ float grpMax16(float v) {
#pragma unroll
  for (int m = 1; m < 16; m <<= 1) v = fmaxf(v, __shfl_xor(v, m, 32));
  return v;
}
__device__ __forceinline__ float grpSum16(float v) {
#pragma unroll
  for (int m = 1; m < 16; m <<= 1) v += __shfl_xor(v, m, 32);
  return v;
}
__device__ __forceinline__ void wait_async0() {
#if __has_builtin(__builtin_amdgcn_s_wait_asynccnt)
  __builtin_amdgcn_s_wait_asynccnt(0);
#else
  asm volatile("s_wait_asynccnt 0x0" ::: "memory");
#endif
}

// ---- async DMA: copy one 256B row-half of bf16 emb into LDS (16 x b128) ----
// GVS form: global_load_async_to_lds_b128 vdst(lds), vaddr(off32), saddr, offset
#define ASYNC_CP16(OFF)                                                        \
  asm volatile("global_load_async_to_lds_b128 %0, %1, %2 offset:" #OFF         \
               :: "v"(lds), "v"(voff), "s"(base) : "memory")

__device__ __forceinline__ void stage_async(unsigned short* tile,
                                            const unsigned short* base,
                                            int n0, int tid) {
  int r = tid >> 1;                       // 0..63 tile row (vocab entry)
  int half = tid & 1;                     // which 128 of 256 K values
  int v = n0 + r; if (v > V_ - 1) v = V_ - 1;
  unsigned lds  = (unsigned)(size_t)(tile + r * TPITCH_ + half * 128);
  unsigned voff = (unsigned)(((size_t)v * D_ + half * 128) * sizeof(unsigned short));
  ASYNC_CP16(0);   ASYNC_CP16(16);  ASYNC_CP16(32);  ASYNC_CP16(48);
  ASYNC_CP16(64);  ASYNC_CP16(80);  ASYNC_CP16(96);  ASYNC_CP16(112);
  ASYNC_CP16(128); ASYNC_CP16(144); ASYNC_CP16(160); ASYNC_CP16(176);
  ASYNC_CP16(192); ASYNC_CP16(208); ASYNC_CP16(224); ASYNC_CP16(240);
}

// ---- fallback staging: f32 global -> pack bf16 -> LDS ----
__device__ __forceinline__ void load_emb_tile(unsigned short* ebt, const float* emb,
                                              int n0, int tid) {
  int r = tid >> 1;
  int halfc = (tid & 1) * 128;
  int v = n0 + r; if (v > V_ - 1) v = V_ - 1;
  const float* src = emb + (size_t)v * D_ + halfc;
  unsigned short* dst = ebt + r * TPITCH_ + halfc;
#pragma unroll
  for (int c = 0; c < 128; c += 8) {
    float4 a = *(const float4*)(src + c);
    float4 b = *(const float4*)(src + c + 4);
    uint4 o;
    o.x = packbf2(a.x, a.y); o.y = packbf2(a.z, a.w);
    o.z = packbf2(b.x, b.y); o.w = packbf2(b.z, b.w);
    *(uint4*)(dst + c) = o;
  }
}

// A operand (16-bit 16x32 layout): lane m holds row m; half selects K chunks.
__device__ __forceinline__ void load_A(BfTile (&A)[2][8], const unsigned short* proj_bf,
                                       int rowbase, int lane) {
  int lane16 = lane & 15, half = lane >> 4;
#pragma unroll
  for (int mt = 0; mt < 2; ++mt) {
    const unsigned short* rp = proj_bf + (size_t)(rowbase + mt * 16 + lane16) * D_;
#pragma unroll
    for (int ks = 0; ks < 8; ++ks) {
      const unsigned short* p = rp + ks * 32 + half * 16;
      A[mt][ks].q[0] = *(const uint4*)(p);
      A[mt][ks].q[1] = *(const uint4*)(p + 8);
    }
  }
}

// 32 rows x 64 cols x K=256 via v_wmma_f32_16x16x32_bf16; B from LDS tile.
__device__ __forceinline__ void compute_tile(f32x8 (&c)[2][4], const BfTile (&A)[2][8],
                                             const unsigned short* ebt, int lane) {
  int lane16 = lane & 15, khalf = (lane >> 4) * 16;
#pragma unroll
  for (int nt = 0; nt < 4; ++nt) {
    const unsigned short* brow = ebt + (nt * 16 + lane16) * TPITCH_ + khalf;
#pragma unroll
    for (int ks = 0; ks < 8; ++ks) {
      BfTile bop;
      const uint4* bp = (const uint4*)(brow + ks * 32);
      bop.q[0] = bp[0]; bop.q[1] = bp[1];
      c[0][nt] = __builtin_amdgcn_wmma_f32_16x16x32_bf16(
          false, A[0][ks].v, false, bop.v, (short)0, c[0][nt], false, false);
      c[1][nt] = __builtin_amdgcn_wmma_f32_16x16x32_bf16(
          false, A[1][ks].v, false, bop.v, (short)0, c[1][nt], false, false);
    }
  }
}

// online-softmax stats update over one 32x64 logits tile
__device__ __forceinline__ void stats_update(float (&mrun)[2][8], float (&srun)[2][8],
                                             const f32x8 (&c)[2][4], int n0, int lane16) {
#pragma unroll
  for (int mt = 0; mt < 2; ++mt) {
#pragma unroll
    for (int r = 0; r < 8; ++r) {
      float vals[4]; float tm = -3.0e38f;
#pragma unroll
      for (int nt = 0; nt < 4; ++nt) {
        int v = n0 + nt * 16 + lane16;
        float xv = (v < V_) ? c[mt][nt][r] : -3.0e38f;
        vals[nt] = xv; tm = fmaxf(tm, xv);
      }
      tm = grpMax16(tm);
      float mn = fmaxf(mrun[mt][r], tm);
      float ts = 0.f;
#pragma unroll
      for (int nt = 0; nt < 4; ++nt) ts += __expf(vals[nt] - mn);  // OOB -> 0
      ts = grpSum16(ts);
      srun[mt][r] = srun[mt][r] * __expf(mrun[mt][r] - mn) + ts;
      mrun[mt][r] = mn;
    }
  }
}

// pass-2 epilogue: normalize, mix the 4 heads, store (exclusive ownership)
__device__ __forceinline__ void mix_store(const f32x8 (&c)[2][4], const float* mrow,
                                          const float* sinv, const float* pi,
                                          float* out, int rowbase, int n0, int lane) {
  int lane16 = lane & 15, half = lane >> 4;
#pragma unroll
  for (int mt = 0; mt < 2; ++mt) {
    float mr[8], si[8], wg[8];
    int rb = rowbase + mt * 16;
#pragma unroll
    for (int r = 0; r < 8; ++r) {            // row M = r + 8*half
      int row = rb + half * 8 + r;
      mr[r] = mrow[row]; si[r] = sinv[row]; wg[r] = pi[row];
    }
    int bA = (rb >> 2) + 2 * half;
#pragma unroll
    for (int nt = 0; nt < 4; ++nt) {
      int v = n0 + nt * 16 + lane16;
      if (v < V_) {
        float pA = 0.f, pB = 0.f;
#pragma unroll
        for (int r = 0; r < 4; ++r) pA += wg[r] * __expf(c[mt][nt][r] - mr[r]) * si[r];
#pragma unroll
        for (int r = 4; r < 8; ++r) pB += wg[r] * __expf(c[mt][nt][r] - mr[r]) * si[r];
        out[(size_t)bA * V_ + v]       = pA;
        out[(size_t)(bA + 1) * V_ + v] = pB;
      }
    }
  }
}

// ---------- kernel 1: proj = tanh(x @ proj_mat^T) -> bf16 [4096 x 256] ----------
__global__ void mos_proj(const float* __restrict__ x, const float* __restrict__ pm,
                         unsigned short* __restrict__ proj_bf) {
  __shared__ float xs[D_];
  int b = blockIdx.x, tid = threadIdx.x;   // 256 threads
  xs[tid] = x[b * D_ + tid];
  __syncthreads();
  int j = blockIdx.y * 256 + tid;
  const float* w = pm + (size_t)j * D_;
  float acc = 0.f;
#pragma unroll 8
  for (int k = 0; k < D_; k += 4) {
    float4 wv = *(const float4*)(w + k);
    acc += wv.x * xs[k] + wv.y * xs[k + 1] + wv.z * xs[k + 2] + wv.w * xs[k + 3];
  }
  proj_bf[(size_t)b * (H_ * D_) + j] = f2bf(tanhf(acc));
}

// ---------- kernel 2: pi = softmax(x @ mix^T), stored flat pi[b*4+h] ----------
__global__ void mos_pi(const float* __restrict__ x, const float* __restrict__ mm,
                       float* __restrict__ pi) {
  int b = blockIdx.x * blockDim.x + threadIdx.x;
  if (b >= B_) return;
  const float* xr = x + (size_t)b * D_;
  float l[H_];
#pragma unroll
  for (int h = 0; h < H_; ++h) {
    const float* w = mm + h * D_;
    float acc = 0.f;
    for (int k = 0; k < D_; k += 4) {
      float4 xv = *(const float4*)(xr + k);
      float4 wv = *(const float4*)(w + k);
      acc += xv.x * wv.x + xv.y * wv.y + xv.z * wv.z + xv.w * wv.w;
    }
    l[h] = acc;
  }
  float m = fmaxf(fmaxf(l[0], l[1]), fmaxf(l[2], l[3]));
  float e0 = __expf(l[0] - m), e1 = __expf(l[1] - m);
  float e2 = __expf(l[2] - m), e3 = __expf(l[3] - m);
  float inv = 1.f / (e0 + e1 + e2 + e3);
  pi[b * H_ + 0] = e0 * inv; pi[b * H_ + 1] = e1 * inv;
  pi[b * H_ + 2] = e2 * inv; pi[b * H_ + 3] = e3 * inv;
}

// ---------- kernel 2b: emb f32 -> bf16 in workspace (fast path) ----------
__global__ void mos_embcvt(const float* __restrict__ emb, unsigned short* __restrict__ ebf) {
  size_t i = ((size_t)blockIdx.x * 256 + threadIdx.x) * 8;
  float4 a = *(const float4*)(emb + i);
  float4 b = *(const float4*)(emb + i + 4);
  uint4 o;
  o.x = packbf2(a.x, a.y); o.y = packbf2(a.z, a.w);
  o.z = packbf2(b.x, b.y); o.w = packbf2(b.z, b.w);
  *(uint4*)(ebf + i) = o;
}

// ---------- kernel 3 (fast): async double-buffered streaming stats ----------
__global__ void __launch_bounds__(128, 1)
mos_pass1_bf(const unsigned short* __restrict__ proj_bf, const unsigned short* __restrict__ ebf,
             float* __restrict__ mpart, float* __restrict__ spart) {
  __shared__ alignas(16) unsigned short tiles[2 * TILE_ELEMS_];
  int tid = threadIdx.x, lane = tid & 31, wave = tid >> 5;
  int rowbase = blockIdx.x * 128 + wave * 32;
  int chunk = blockIdx.y;
  int lane16 = lane & 15;

  BfTile A[2][8];
  load_A(A, proj_bf, rowbase, lane);

  float mrun[2][8], srun[2][8];
#pragma unroll
  for (int mt = 0; mt < 2; ++mt)
#pragma unroll
    for (int r = 0; r < 8; ++r) { mrun[mt][r] = -3.0e38f; srun[mt][r] = 0.0f; }

  // prologue: DMA tile 0
  stage_async(tiles, ebf, chunk * IPC_ * 64, tid);

  for (int i = 0; i < IPC_; ++i) {
    int iter = chunk * IPC_ + i;
    if (iter >= NITER_) break;        // uniform across block
    int n0 = iter * 64;
    wait_async0();                    // own DMA for tile i complete
    __syncthreads();                  // all waves: tile i in LDS, tile (i-1) reads consumed
    // issue DMA for tile i+1 into the other buffer, overlapping the WMMA below
    if (i + 1 < IPC_ && iter + 1 < NITER_)
      stage_async(tiles + ((i + 1) & 1) * TILE_ELEMS_, ebf, n0 + 64, tid);

    f32x8 c[2][4] = {};
    compute_tile(c, A, tiles + (i & 1) * TILE_ELEMS_, lane);
    stats_update(mrun, srun, c, n0, lane16);
  }

  if (lane16 == 0) {
    int half = lane >> 4;
#pragma unroll
    for (int mt = 0; mt < 2; ++mt)
#pragma unroll
      for (int r = 0; r < 8; ++r) {
        int row = rowbase + mt * 16 + half * 8 + r;
        mpart[chunk * ROWS_ + row] = mrun[mt][r];
        spart[chunk * ROWS_ + row] = srun[mt][r];
      }
  }
}

// ---------- kernel 3 (fallback): pack-on-stage streaming stats ----------
__global__ void __launch_bounds__(128, 1)
mos_pass1(const unsigned short* __restrict__ proj_bf, const float* __restrict__ emb,
          float* __restrict__ mpart, float* __restrict__ spart) {
  __shared__ alignas(16) unsigned short ebt[TILE_ELEMS_];
  int tid = threadIdx.x, lane = tid & 31, wave = tid >> 5;
  int rowbase = blockIdx.x * 128 + wave * 32;
  int chunk = blockIdx.y;
  int lane16 = lane & 15;

  BfTile A[2][8];
  load_A(A, proj_bf, rowbase, lane);

  float mrun[2][8], srun[2][8];
#pragma unroll
  for (int mt = 0; mt < 2; ++mt)
#pragma unroll
    for (int r = 0; r < 8; ++r) { mrun[mt][r] = -3.0e38f; srun[mt][r] = 0.0f; }

  for (int i = 0; i < IPC_; ++i) {
    int iter = chunk * IPC_ + i;
    if (iter >= NITER_) break;
    int n0 = iter * 64;
    __syncthreads();
    load_emb_tile(ebt, emb, n0, tid);
    __syncthreads();

    f32x8 c[2][4] = {};
    compute_tile(c, A, ebt, lane);

    if (iter + 1 < NITER_) {          // prefetch next tile (global_prefetch_b8)
      int vn = n0 + 64 + (tid >> 1); if (vn > V_ - 1) vn = V_ - 1;
      __builtin_prefetch(emb + (size_t)vn * D_ + (tid & 1) * 128, 0, 1);
    }
    stats_update(mrun, srun, c, n0, lane16);
  }

  if (lane16 == 0) {
    int half = lane >> 4;
#pragma unroll
    for (int mt = 0; mt < 2; ++mt)
#pragma unroll
      for (int r = 0; r < 8; ++r) {
        int row = rowbase + mt * 16 + half * 8 + r;
        mpart[chunk * ROWS_ + row] = mrun[mt][r];
        spart[chunk * ROWS_ + row] = srun[mt][r];
      }
  }
}

// ---------- kernel 4: merge chunk stats (log-sum-exp) ----------
__global__ void mos_merge(const float* __restrict__ mpart, const float* __restrict__ spart,
                          float* __restrict__ mrow, float* __restrict__ sinv) {
  int row = blockIdx.x * blockDim.x + threadIdx.x;
  if (row >= ROWS_) return;
  float m = -3.0e38f;
#pragma unroll
  for (int c = 0; c < NCHUNK_; ++c) m = fmaxf(m, mpart[c * ROWS_ + row]);
  float s = 0.f;
#pragma unroll
  for (int c = 0; c < NCHUNK_; ++c)
    s += spart[c * ROWS_ + row] * __expf(mpart[c * ROWS_ + row] - m);
  mrow[row] = m;
  sinv[row] = 1.0f / s;
}

// ---------- kernel 5 (fast): async-staged recompute + mix + store ----------
__global__ void __launch_bounds__(128, 1)
mos_pass2_bf(const unsigned short* __restrict__ proj_bf, const unsigned short* __restrict__ ebf,
             const float* __restrict__ pi, const float* __restrict__ mrow,
             const float* __restrict__ sinv, float* __restrict__ out) {
  __shared__ alignas(16) unsigned short ebt[TILE_ELEMS_];
  int tid = threadIdx.x, lane = tid & 31, wave = tid >> 5;
  int rowbase = blockIdx.x * 128 + wave * 32;
  int n0 = blockIdx.y * 64;

  stage_async(ebt, ebf, n0, tid);     // DMA tile while A loads proceed
  BfTile A[2][8];
  load_A(A, proj_bf, rowbase, lane);
  wait_async0();
  __syncthreads();

  f32x8 c[2][4] = {};
  compute_tile(c, A, ebt, lane);
  mix_store(c, mrow, sinv, pi, out, rowbase, n0, lane);
}

// ---------- kernel 5 (fallback) ----------
__global__ void __launch_bounds__(128, 1)
mos_pass2(const unsigned short* __restrict__ proj_bf, const float* __restrict__ emb,
          const float* __restrict__ pi, const float* __restrict__ mrow,
          const float* __restrict__ sinv, float* __restrict__ out) {
  __shared__ alignas(16) unsigned short ebt[TILE_ELEMS_];
  int tid = threadIdx.x, lane = tid & 31, wave = tid >> 5;
  int rowbase = blockIdx.x * 128 + wave * 32;
  int n0 = blockIdx.y * 64;

  load_emb_tile(ebt, emb, n0, tid);
  BfTile A[2][8];
  load_A(A, proj_bf, rowbase, lane);
  __syncthreads();

  f32x8 c[2][4] = {};
  compute_tile(c, A, ebt, lane);
  mix_store(c, mrow, sinv, pi, out, rowbase, n0, lane);
}

// ---------- launch ----------
extern "C" void kernel_launch(void* const* d_in, const int* in_sizes, int n_in,
                              void* d_out, int out_size, void* d_ws, size_t ws_size,
                              hipStream_t stream) {
  const float* x   = (const float*)d_in[0];
  const float* pm  = (const float*)d_in[1];
  const float* mm  = (const float*)d_in[2];
  const float* emb = (const float*)d_in[3];
  float* out = (float*)d_out;

  char* ws = (char*)d_ws;
  unsigned short* proj_bf = (unsigned short*)(ws);
  size_t off = (size_t)ROWS_ * D_ * sizeof(unsigned short);          // 2 MiB
  float* pi    = (float*)(ws + off); off += (size_t)ROWS_ * sizeof(float);
  float* mpart = (float*)(ws + off); off += (size_t)NCHUNK_ * ROWS_ * sizeof(float);
  float* spart = (float*)(ws + off); off += (size_t)NCHUNK_ * ROWS_ * sizeof(float);
  float* mrowb = (float*)(ws + off); off += (size_t)ROWS_ * sizeof(float);
  float* sinvb = (float*)(ws + off); off += (size_t)ROWS_ * sizeof(float);
  off = (off + 255) & ~(size_t)255;
  unsigned short* embbf = (unsigned short*)(ws + off);
  size_t need = off + (size_t)V_ * D_ * sizeof(unsigned short);      // ~54 MB
  bool fast = (ws_size >= need);

  mos_proj <<<dim3(B_, (H_ * D_) / 256), 256, 0, stream>>>(x, pm, proj_bf);
  mos_pi   <<<dim3(B_ / 128), 128, 0, stream>>>(x, mm, pi);
  if (fast) {
    mos_embcvt  <<<dim3((V_ * D_) / (256 * 8)), 256, 0, stream>>>(emb, embbf);
    mos_pass1_bf<<<dim3(ROWS_ / 128, NCHUNK_), 128, 0, stream>>>(proj_bf, embbf, mpart, spart);
  } else {
    mos_pass1   <<<dim3(ROWS_ / 128, NCHUNK_), 128, 0, stream>>>(proj_bf, emb, mpart, spart);
  }
  mos_merge<<<dim3(ROWS_ / 256), 256, 0, stream>>>(mpart, spart, mrowb, sinvb);
  if (fast) {
    mos_pass2_bf<<<dim3(ROWS_ / 128, NITER_), 128, 0, stream>>>(proj_bf, embbf, pi, mrowb, sinvb, out);
  } else {
    mos_pass2   <<<dim3(ROWS_ / 128, NITER_), 128, 0, stream>>>(proj_bf, emb, pi, mrowb, sinvb, out);
  }
}
